// Conv_88905823027894
// MI455X (gfx1250) — compile-verified, matrix operand order checked by reading, not run on hardware
//
#include <hip/hip_runtime.h>

// ---------------------------------------------------------------------------
// Differentiable logic-gate conv net, fused for MI455X (gfx1250, wave32).
//
// Kernel 1 (coef_wmma_kernel): hoists softmax(w) @ GATE_M out of the spatial
//   loop. (4032 gates x 16 probs) x (16 x 4 GATE_M) done as 252 wave-tiles of
//   V_WMMA_F32_16X16X32_F16 (A = probs f16 zero-padded to K=32, B = GATE_M
//   zero-padded to 32x16). Writes 4032 x float4 coefficients into d_ws.
//
// Kernel 2 (lgn_conv_kernel): fully fused forward. Per block: 4 spatial rows,
//   im2col patches in LDS, one thread per (row, o-channel) runs all 6 logic
//   layers in a private 32-float LDS slot. Only 12 MB of HBM traffic total.
// ---------------------------------------------------------------------------

typedef __attribute__((ext_vector_type(16))) _Float16 v16h;
typedef __attribute__((ext_vector_type(8)))  float    v8f;

#define N_ROWS      32768      // B * OH * OW = 32*32*32
#define TOTAL_GATES 4032       // 64 * (32+16+8+4+2+1)
// flattened gate-layer base offsets (layer-major, [l][o][j])
#define G0 0
#define G1 2048
#define G2 3072
#define G3 3584
#define G4 3840
#define G5 3968

__constant__ float GATE_M[16][4] = {
    {0,0,0,0},   {0,0,0,1},   {0,1,0,-1},  {0,1,0,0},
    {0,0,1,-1},  {0,0,1,0},   {0,1,1,-2},  {0,1,1,-1},
    {1,-1,-1,1}, {1,-1,-1,2}, {1,0,-1,0},  {1,0,-1,1},
    {1,-1,0,0},  {1,-1,0,1},  {1,0,0,-1},  {1,0,0,0}};

// ---------------------------------------------------------------------------
// Kernel 1: coef[g][0..3] = softmax(w[g][0..15]) @ GATE_M, via WMMA.
// One wave32 block per 16-gate tile; 252 blocks cover all 4032 gates.
// ---------------------------------------------------------------------------
__global__ __launch_bounds__(32)
void coef_wmma_kernel(const float* __restrict__ w0, const float* __restrict__ w1,
                      const float* __restrict__ w2, const float* __restrict__ w3,
                      const float* __restrict__ w4, const float* __restrict__ w5,
                      float* __restrict__ coef /* [4032][4] */)
{
    __shared__ float probs[16][16];
    const int lane = threadIdx.x;
    const int g0   = blockIdx.x * 16;

    // Which layer does this tile belong to? (layer sizes are multiples of 16)
    const float* w; int base;
    if      (g0 < G1) { w = w0; base = G0; }
    else if (g0 < G2) { w = w1; base = G1; }
    else if (g0 < G3) { w = w2; base = G2; }
    else if (g0 < G4) { w = w3; base = G3; }
    else if (g0 < G5) { w = w4; base = G4; }
    else              { w = w5; base = G5; }

    // Softmax over 16 logits: one gate-row per lane 0..15.
    if (lane < 16) {
        const float* wr = w + (size_t)(g0 - base + lane) * 16;
        float m = wr[0];
        #pragma unroll
        for (int k = 1; k < 16; ++k) m = fmaxf(m, wr[k]);
        float e[16], s = 0.0f;
        #pragma unroll
        for (int k = 0; k < 16; ++k) { e[k] = __expf(wr[k] - m); s += e[k]; }
        const float inv = 1.0f / s;
        #pragma unroll
        for (int k = 0; k < 16; ++k) probs[lane][k] = e[k] * inv;
    }
    __syncthreads();

    // Pack A (16x32 f16): lane L and L+16 both hold row L; lane L carries
    // K=0..7 (halfs 0-7) + K=16..23 (halfs 8-15, zero-pad); lane L+16 carries
    // K=8..15 + K=24..31 (zero-pad).  (ISA 7.12.2, 16-bit A layout)
    const int row = lane & 15;
    const int hiK = (lane >> 4) & 1;
    v16h A, Bm;
    #pragma unroll
    for (int h = 0; h < 8; ++h) {
        A[h]     = (_Float16)probs[row][hiK * 8 + h];
        A[h + 8] = (_Float16)0.0f;
    }
    // Pack B (32x16 f16): B[k][n] = GATE_M[k][n] for k<16, n<4; zero elsewhere.
    const int col = lane & 15;
    #pragma unroll
    for (int h = 0; h < 8; ++h) {
        float bv = (col < 4) ? GATE_M[hiK * 8 + h][col] : 0.0f;
        Bm[h]     = (_Float16)bv;
        Bm[h + 8] = (_Float16)0.0f;
    }
    v8f C;
    #pragma unroll
    for (int i = 0; i < 8; ++i) C[i] = 0.0f;

    // D(16x16) = A(16x32) x B(32x16) + 0   -> v_wmma_f32_16x16x32_f16
    v8f D = __builtin_amdgcn_wmma_f32_16x16x32_f16(
        /*neg_a=*/false, A, /*neg_b=*/false, Bm,
        /*c_mod=*/(short)0, C, /*reuse_a=*/false, /*reuse_b=*/false);

    // D layout: lanes 0-15 -> N=lane, M=v; lanes 16-31 -> N=lane-16, M=v+8.
    const int n     = lane & 15;
    const int mbase = (lane < 16) ? 0 : 8;
    if (n < 4) {
        #pragma unroll
        for (int v = 0; v < 8; ++v)
            coef[(size_t)(g0 + mbase + v) * 4 + n] = D[v];
    }
}

// ---------------------------------------------------------------------------
// Kernel 2: fused im2col + 6 logic layers + output transpose.
// ---------------------------------------------------------------------------
#define ROWS_PER_BLOCK 4
#define THREADS        256
#define HSTRIDE        33   // pad 32->33 floats to break LDS bank conflicts

template<int DO>
__device__ __forceinline__ void logic_layer(const int* __restrict__ ia,
                                            const int* __restrict__ ib,
                                            const float4* __restrict__ cf,
                                            int base, int o,
                                            float* __restrict__ h)
{
    float tmp[DO];
    #pragma unroll
    for (int g = 0; g < DO; ++g) {
        const int gi = o * DO + g;
        const float a = h[ia[gi]];
        const float b = h[ib[gi]];
        const float4 c = cf[base + gi];
        tmp[g] = fmaf(c.w, a * b, fmaf(c.z, b, fmaf(c.y, a, c.x)));
    }
    #pragma unroll
    for (int g = 0; g < DO; ++g) h[g] = tmp[g];
}

__global__ __launch_bounds__(THREADS)
void lgn_conv_kernel(const float* __restrict__ x,
                     const int* __restrict__ ia0, const int* __restrict__ ib0,
                     const int* __restrict__ ia1, const int* __restrict__ ib1,
                     const int* __restrict__ ia2, const int* __restrict__ ib2,
                     const int* __restrict__ ia3, const int* __restrict__ ib3,
                     const int* __restrict__ ia4, const int* __restrict__ ib4,
                     const int* __restrict__ ia5, const int* __restrict__ ib5,
                     const float4* __restrict__ cf,
                     float* __restrict__ out)
{
    __shared__ float s_pat[ROWS_PER_BLOCK][288];   //  4.5 KB: im2col patches
    __shared__ float s_h[THREADS * HSTRIDE];       // 33.0 KB: per-thread hidden

    const int t  = threadIdx.x;
    const int n0 = blockIdx.x * ROWS_PER_BLOCK;

    // Cooperative im2col: patch[p] = x[b, c, i+ki-1, j+kj-1], p = c*9+ki*3+kj.
    for (int e = t; e < ROWS_PER_BLOCK * 288; e += THREADS) {
        const int rr = e / 288, p  = e - rr * 288;
        const int c  = p / 9,   s  = p - c * 9;
        const int ki = s / 3 - 1;
        const int kj = s - (s / 3) * 3 - 1;
        const int nn = n0 + rr;
        const int bb = nn >> 10, pp = nn & 1023;
        const int yi = (pp >> 5) + ki;
        const int xj = (pp & 31) + kj;
        float v = 0.0f;
        if ((unsigned)yi < 32u && (unsigned)xj < 32u)
            v = x[((((bb << 5) + c) << 5) + yi) * 32 + xj];
        s_pat[rr][p] = v;
    }
    __syncthreads();

    const int r = t >> 6;          // spatial row within block (0..3)
    const int o = t & 63;          // output channel (0..63)
    float* __restrict__ h  = &s_h[t * HSTRIDE];
    const float* __restrict__ pat = s_pat[r];

    // Layer 0: 288 -> 32 (gathers from the shared patch row).
    #pragma unroll 4
    for (int g = 0; g < 32; ++g) {
        const int gi = (o << 5) + g;
        const float a = pat[ia0[gi]];
        const float b = pat[ib0[gi]];
        const float4 c = cf[G0 + gi];
        h[g] = fmaf(c.w, a * b, fmaf(c.z, b, fmaf(c.y, a, c.x)));
    }
    // Layers 1..5 operate on the thread's private hidden vector.
    logic_layer<16>(ia1, ib1, cf, G1, o, h);
    logic_layer< 8>(ia2, ib2, cf, G2, o, h);
    logic_layer< 4>(ia3, ib3, cf, G3, o, h);
    logic_layer< 2>(ia4, ib4, cf, G4, o, h);
    logic_layer< 1>(ia5, ib5, cf, G5, o, h);

    // y[b, o, i, j] = h[n, o],  n = b*1024 + (i*32 + j)
    const int nrow = n0 + r;
    const int b    = nrow >> 10, pos = nrow & 1023;
    out[(((b << 6) + o) << 10) + pos] = h[0];
}

// ---------------------------------------------------------------------------
extern "C" void kernel_launch(void* const* d_in, const int* in_sizes, int n_in,
                              void* d_out, int out_size, void* d_ws, size_t ws_size,
                              hipStream_t stream)
{
    (void)in_sizes; (void)n_in; (void)out_size; (void)ws_size;
    // dict order: x, then per layer l: ia{l}, ib{l}, w{l}
    const float* x = (const float*)d_in[0];
    const int*   ia[6]; const int* ib[6]; const float* w[6];
    for (int l = 0; l < 6; ++l) {
        ia[l] = (const int*)  d_in[1 + 3 * l];
        ib[l] = (const int*)  d_in[2 + 3 * l];
        w[l]  = (const float*)d_in[3 + 3 * l];
    }
    float* coef = (float*)d_ws;   // 4032 * 4 floats = 64512 B of scratch

    coef_wmma_kernel<<<TOTAL_GATES / 16, 32, 0, stream>>>(
        w[0], w[1], w[2], w[3], w[4], w[5], coef);

    lgn_conv_kernel<<<N_ROWS / ROWS_PER_BLOCK, THREADS, 0, stream>>>(
        x,
        ia[0], ib[0], ia[1], ib[1], ia[2], ib[2],
        ia[3], ib[3], ia[4], ib[4], ia[5], ib[5],
        (const float4*)d_ws, (float*)d_out);
}
